// CausalSelfAttention_TSSA_38199439130941
// MI455X (gfx1250) — compile-verified, matrix operand order checked by reading, not run on hardware
//
#include <hip/hip_runtime.h>
#include <cstdint>
#include <cstddef>

// ---------- problem constants (B=4, T=4096, C=2048, H=16, D=128) ----------
#define BB 4
#define TT 4096
#define CC 2048
#define HH 16
#define DD 128

typedef __attribute__((ext_vector_type(16))) __bf16    v16bf;
typedef __attribute__((ext_vector_type(8)))  float     v8f;
typedef __attribute__((ext_vector_type(4)))  float     f4;
typedef __attribute__((ext_vector_type(4)))  unsigned  u32x4;

// ---------- CDNA5 async copy (global -> LDS), tracked by ASYNCcnt ----------
__device__ __forceinline__ void async_ld16(unsigned ldsOff, unsigned long long gaddr) {
    asm volatile("global_load_async_to_lds_b128 %0, %1, off"
                 :: "v"(ldsOff), "v"(gaddr) : "memory");
}
__device__ __forceinline__ void wait_async_le4() { asm volatile("s_wait_asynccnt 4" ::: "memory"); }
__device__ __forceinline__ void wait_async_0()  { asm volatile("s_wait_asynccnt 0" ::: "memory"); }

// round-to-nearest-even f32 -> bf16 (bit pattern), host-safe kernel signatures
__device__ __forceinline__ unsigned short f2bf(float f) {
    unsigned u = __float_as_uint(f);
    unsigned r = 0x7fffu + ((u >> 16) & 1u);
    return (unsigned short)((u + r) >> 16);
}

// ---------- elementwise f32 -> bf16 conversion ----------
__global__ void cvt_f32_bf16(const float* __restrict__ in, unsigned short* __restrict__ out,
                             long long n) {
    long long i = (long long)blockIdx.x * blockDim.x + threadIdx.x;
    long long stride = (long long)gridDim.x * blockDim.x;
    for (; i < n; i += stride) out[i] = f2bf(in[i]);
}

// ---------- bf16 WMMA GEMM: Cout[M,N] = A[M,K] * Bw[N,K]^T (both K-major) ----------
// block tile 128(M) x 128(N); 256 threads = 8 waves in 4x2; wave tile 32x64
// K staged 32 at a time in LDS, double-buffered async copies.
__global__ __launch_bounds__(256) void gemm_bf16_wmma(
        const unsigned short* __restrict__ A,
        const unsigned short* __restrict__ Bw,
        float* __restrict__ Cout, int M, int N, int K)
{
    __shared__ __align__(128) unsigned char smem[2 * 16384]; // 2 x (8KB A + 8KB B)
    const int tid   = threadIdx.x;
    const int lane  = tid & 31;
    const int wave  = tid >> 5;
    const int l15   = lane & 15;
    const int kbB   = (lane < 16) ? 0 : 16;    // byte offset of this half-lane's first 8 elems
    const int m_off = (wave & 3) * 32;
    const int n_off = (wave >> 2) * 64;
    const int mBase = blockIdx.y * 128;
    const int nBase = blockIdx.x * 128;
    const unsigned long long Abytes = (unsigned long long)(uintptr_t)A;
    const unsigned long long Bbytes = (unsigned long long)(uintptr_t)Bw;
    const int KT = K >> 5;                      // K / 32

    // Escape the LDS array's address into opaque asm: from here on, every
    // asm volatile with a "memory" clobber (async copies / waits) must be
    // assumed to write smem, so DS loads below can't be folded or CSE'd.
    {
        const void* esc = (const void*)smem;
        asm volatile("" :: "v"(esc) : "memory");
    }
    // Priming store at an opaque offset: the LDS object is visibly written.
    smem[(unsigned)(tid * 129) & 32767u] = (unsigned char)tid;

    auto issue_stage = [&](int kt, int buf) {
        const unsigned bufOff = (unsigned)buf * 16384u;
        const unsigned long long kByte = (unsigned long long)kt * 64ull;
        #pragma unroll
        for (int i = 0; i < 2; ++i) {
            unsigned c     = (unsigned)(i * 256 + tid);
            unsigned off   = c * 16u;           // byte offset inside 8KB tile
            unsigned row   = off >> 6;          // 64 bytes per row (32 bf16)
            unsigned inrow = off & 63u;
            async_ld16(bufOff + off,
                Abytes + ((unsigned long long)(mBase + (int)row) * (unsigned long long)K) * 2ull
                       + kByte + inrow);
            async_ld16(bufOff + 8192u + off,
                Bbytes + ((unsigned long long)(nBase + (int)row) * (unsigned long long)K) * 2ull
                       + kByte + inrow);
        }
    };

    const v8f vzero = {0.f, 0.f, 0.f, 0.f, 0.f, 0.f, 0.f, 0.f};
    v8f acc[2][4];
    #pragma unroll
    for (int i = 0; i < 2; ++i)
        #pragma unroll
        for (int j = 0; j < 4; ++j) acc[i][j] = vzero;

    issue_stage(0, 0);

    for (int kt = 0; kt < KT; ++kt) {
        const int buf = kt & 1;
        if (kt + 1 < KT) { issue_stage(kt + 1, buf ^ 1); wait_async_le4(); }
        else             { wait_async_0(); }
        __syncthreads();

        const unsigned bufOff = (unsigned)buf * 16384u;
        union Frag { u32x4 q[2]; v16bf v; };
        Frag fa[2], fb[4];
        // plain addrspace(3) reads -> ds_load_b128, waited via DScnt
        #pragma unroll
        for (int i = 0; i < 2; ++i) {
            unsigned base = bufOff + (unsigned)(m_off + i * 16 + l15) * 64u + (unsigned)kbB;
            fa[i].q[0] = *(const u32x4*)&smem[base];
            fa[i].q[1] = *(const u32x4*)&smem[base + 32u];
        }
        #pragma unroll
        for (int j = 0; j < 4; ++j) {
            unsigned base = bufOff + 8192u + (unsigned)(n_off + j * 16 + l15) * 64u + (unsigned)kbB;
            fb[j].q[0] = *(const u32x4*)&smem[base];
            fb[j].q[1] = *(const u32x4*)&smem[base + 32u];
        }
        #pragma unroll
        for (int i = 0; i < 2; ++i)
            #pragma unroll
            for (int j = 0; j < 4; ++j)
                acc[i][j] = __builtin_amdgcn_wmma_f32_16x16x32_bf16(
                    false, fa[i].v, false, fb[j].v, (short)0, acc[i][j], false, false);
        __syncthreads();
    }

    // C/D layout: VGPR r -> M=r (lanes 0-15) / M=8+r (lanes 16-31); N = lane&15
    const int rbase = (lane < 16) ? 0 : 8;
    #pragma unroll
    for (int i = 0; i < 2; ++i)
        #pragma unroll
        for (int j = 0; j < 4; ++j)
            #pragma unroll
            for (int r = 0; r < 8; ++r) {
                size_t row = (size_t)(mBase + m_off + i * 16 + rbase + r);
                size_t col = (size_t)(nBase + n_off + j * 16 + l15);
                Cout[row * (size_t)N + col] = acc[i][j][r];
            }
}

// ---------- scan 1: tmp[b,h,t] = (sum_d w^2/max(cumsum w^2,1e-12) + D*bias) * temp[h] ----------
// one wave per (b,h); lane owns 4 contiguous d's (float4)
__global__ __launch_bounds__(32) void tssa_tmp_kernel(
        const float* __restrict__ w, const float* __restrict__ temp,
        const float* __restrict__ bias, float* __restrict__ tmp)
{
    const int b = blockIdx.x / HH, h = blockIdx.x % HH;
    const int lane = threadIdx.x;
    const float th = temp[h];
    const float* wp = w + (size_t)b * TT * CC + h * DD + lane * 4;
    const float* bp = bias + (size_t)h * TT;
    float* op = tmp + ((size_t)b * HH + h) * TT;
    float d0 = 0.f, d1 = 0.f, d2 = 0.f, d3 = 0.f;
    for (int t = 0; t < TT; ++t) {
        f4 wv = *(const f4*)wp;
        wp += CC;
        float v0 = wv[0] * wv[0], v1 = wv[1] * wv[1], v2 = wv[2] * wv[2], v3 = wv[3] * wv[3];
        d0 += v0; d1 += v1; d2 += v2; d3 += v3;
        float s = v0 / fmaxf(d0, 1e-12f) + v1 / fmaxf(d1, 1e-12f)
                + v2 / fmaxf(d2, 1e-12f) + v3 / fmaxf(d3, 1e-12f);
        #pragma unroll
        for (int off = 16; off > 0; off >>= 1) s += __shfl_xor(s, off, 32);
        if (lane == 0) op[t] = (s + (float)DD * bp[t]) * th;
    }
}

// ---------- softmax over heads: Pi[b,h,t] = softmax_h(tmp[b,h,t]) ----------
__global__ void tssa_pi_kernel(const float* __restrict__ tmp, float* __restrict__ Pi) {
    int idx = blockIdx.x * blockDim.x + threadIdx.x;   // over B*T
    if (idx >= BB * TT) return;
    int b = idx / TT, t = idx % TT;
    const float* tp = tmp + (size_t)b * HH * TT + t;
    float vals[HH], m = -3.4e38f;
    #pragma unroll
    for (int h = 0; h < HH; ++h) { vals[h] = tp[(size_t)h * TT]; m = fmaxf(m, vals[h]); }
    float se = 0.f;
    #pragma unroll
    for (int h = 0; h < HH; ++h) { vals[h] = __expf(vals[h] - m); se += vals[h]; }
    float inv = 1.f / se;
    float* pp = Pi + (size_t)b * HH * TT + t;
    #pragma unroll
    for (int h = 0; h < HH; ++h) pp[(size_t)h * TT] = vals[h] * inv;
}

// ---------- scan 2: y = -(w*Pi) / (1 + cumsum(w^2*Pi)/(cumsum(Pi)+1e-8)), write bf16 ----------
__global__ __launch_bounds__(32) void tssa_y_kernel(
        const float* __restrict__ w, const float* __restrict__ Pi,
        unsigned short* __restrict__ yb)
{
    const int b = blockIdx.x / HH, h = blockIdx.x % HH;
    const int lane = threadIdx.x;
    const float* wp = w + (size_t)b * TT * CC + h * DD + lane * 4;
    const float* pp = Pi + ((size_t)b * HH + h) * TT;
    unsigned short* yp = yb + (size_t)b * TT * CC + h * DD + lane * 4;
    float c0 = 0.f, c1 = 0.f, c2 = 0.f, c3 = 0.f, cpi = 0.f;
    for (int t = 0; t < TT; ++t) {
        float pi = pp[t];
        f4 wv = *(const f4*)wp;
        c0 += wv[0] * wv[0] * pi; c1 += wv[1] * wv[1] * pi;
        c2 += wv[2] * wv[2] * pi; c3 += wv[3] * wv[3] * pi;
        cpi += pi;
        float rin = 1.f / (cpi + 1e-8f);
        float y0 = -(wv[0] * pi) / (1.f + c0 * rin);
        float y1 = -(wv[1] * pi) / (1.f + c1 * rin);
        float y2 = -(wv[2] * pi) / (1.f + c2 * rin);
        float y3 = -(wv[3] * pi) / (1.f + c3 * rin);
        typedef __attribute__((ext_vector_type(4))) unsigned short u16x4;
        u16x4 o = { f2bf(y0), f2bf(y1), f2bf(y2), f2bf(y3) };
        *(u16x4*)yp = o;
        wp += CC; yp += CC;
    }
}

// ---------- launch ----------
extern "C" void kernel_launch(void* const* d_in, const int* in_sizes, int n_in,
                              void* d_out, int out_size, void* d_ws, size_t ws_size,
                              hipStream_t stream) {
    (void)in_sizes; (void)n_in; (void)out_size; (void)ws_size;
    const float* x    = (const float*)d_in[0];
    const float* Wa   = (const float*)d_in[1];
    const float* Wp   = (const float*)d_in[2];
    const float* temp = (const float*)d_in[3];
    const float* bias = (const float*)d_in[4];
    float* out = (float*)d_out;

    char* ws = (char*)d_ws;
    // workspace layout (bytes)
    unsigned short* xb  = (unsigned short*)(ws);                 // 67,108,864  (reused as yb)
    unsigned short* Wab = (unsigned short*)(ws +  67108864ull);  //  8,388,608
    unsigned short* Wpb = (unsigned short*)(ws +  75497472ull);  //  8,388,608
    float*          w   = (float*)        (ws +  83886080ull);   // 134,217,728
    float*          tmp = (float*)        (ws + 218103808ull);   //   1,048,576
    float*          Pi  = (float*)        (ws + 219152384ull);   //   1,048,576

    const long long nX = (long long)BB * TT * CC;
    const long long nW = (long long)CC * CC;
    cvt_f32_bf16<<<2048, 256, 0, stream>>>(x,  xb,  nX);
    cvt_f32_bf16<<<512,  256, 0, stream>>>(Wa, Wab, nW);
    cvt_f32_bf16<<<512,  256, 0, stream>>>(Wp, Wpb, nW);

    dim3 gG(CC / 128, (BB * TT) / 128);   // (16, 128)
    gemm_bf16_wmma<<<gG, 256, 0, stream>>>(xb, Wab, w, BB * TT, CC, CC);

    tssa_tmp_kernel<<<BB * HH, 32, 0, stream>>>(w, temp, bias, tmp);
    tssa_pi_kernel<<<(BB * TT + 255) / 256, 256, 0, stream>>>(tmp, Pi);
    tssa_y_kernel<<<BB * HH, 32, 0, stream>>>(w, Pi, xb /* = yb */);

    gemm_bf16_wmma<<<gG, 256, 0, stream>>>(xb, Wpb, out, BB * TT, CC, CC);
}
